// TurboQuant_75539884802800
// MI455X (gfx1250) — matrix operand order repaired and checked by reading, not compiled, not run on hardware
//
#include <hip/hip_runtime.h>
#include <hip/hip_bf16.h>

typedef __attribute__((ext_vector_type(16))) _Float16 v16h;
typedef __attribute__((ext_vector_type(8)))  _Float16 v8h;
typedef __attribute__((ext_vector_type(8)))  float    v8f;
typedef __attribute__((ext_vector_type(4)))  float    v4f;

#define WAVES_PER_WG 8

static __device__ __forceinline__ v8f wmma_f16(v16h a, v16h b, v8f c) {
    // D = A(16x32 f16) * B(32x16 f16) + C(16x16 f32)
    return __builtin_amdgcn_wmma_f32_16x16x32_f16(
        /*neg_a=*/false, a, /*neg_b=*/false, b,
        /*c_mod=*/(short)0, c, /*reuse_a=*/false, /*reuse_b=*/false);
}

__global__ __launch_bounds__(256, 1) void turboquant_kernel(
    const float* __restrict__ x,     // [nblk*16, 64]
    const float* __restrict__ cbG,   // [16] sorted Lloyd-Max codebook
    const float* __restrict__ R,     // [64, 64] row-major rotation
    float* __restrict__ out,         // [nblk*16, 64]
    int nblk)
{
    __shared__ __align__(16) float     Rt[64 * 64];                 // R transposed (16 KB)
    __shared__ __align__(16) _Float16  zt[WAVES_PER_WG][16 * 64];   // per-wave z_hat tile (16 KB)
    __shared__ __align__(16) float     sc[WAVES_PER_WG][16];        // per-wave row scales

    const int tid  = threadIdx.x;
    const int lane = tid & 31;
    const int wave = tid >> 5;
    const int nsel = lane & 15;   // A-row / B-col / D-col selector
    const int hi   = lane >> 4;   // half-wave select
    const int kbA  = hi * 8;      // A-fragment K sub-offset (per ISA 16-bit A layout)

    // ---- one-time: transpose R into LDS (coalesced global reads) ----
    #pragma unroll
    for (int i = 0; i < 16; ++i) {
        int idx = tid + i * 256;
        int r = idx >> 6, c = idx & 63;
        Rt[c * 64 + r] = R[idx];
    }
    __syncthreads();

    // ---- one-time: codebook + interval midpoints (wave-uniform -> SGPRs) ----
    float cb[16], mid[15];
    #pragma unroll
    for (int i = 0; i < 16; ++i) cb[i] = cbG[i];
    #pragma unroll
    for (int i = 0; i < 15; ++i) mid[i] = 0.5f * (cb[i] + cb[i + 1]);

    // ---- one-time: B fragments held in registers for the whole kernel ----
    // B layout (32x16 f16): lane%16 = column N; K table order: lanes 0-15 hold
    // K=k0+{0..7,16..23}, lanes 16-31 hold K=k0+{8..15,24..31} (per-fragment k0 = kh*32).
    // B1 = R^T  (B1[k][n] = R[n][k]) : row reads from global R
    // B2 = R    (B2[k][n] = Rt[n][k]): row reads from LDS Rt
    v16h B1[4][2], B2[4][2];
    #pragma unroll
    for (int nt = 0; nt < 4; ++nt) {
        const int n_g = nt * 16 + nsel;
        #pragma unroll
        for (int kh = 0; kh < 2; ++kh) {
            const int k0 = kh * 32 + hi * 8;
            const float* pg = R  + n_g * 64 + k0;
            const float* pl = Rt + n_g * 64 + k0;
            #pragma unroll
            for (int c = 0; c < 4; ++c) {
                v4f tg = *(const v4f*)(pg + 4 * c + (c & 2) * 4);
                v4f tl = *(const v4f*)(pl + 4 * c + (c & 2) * 4);
                #pragma unroll
                for (int j = 0; j < 4; ++j) {
                    B1[nt][kh][4 * c + j] = (_Float16)tg[j];
                    B2[nt][kh][4 * c + j] = (_Float16)tl[j];
                }
            }
        }
    }

    const int wave_gid = blockIdx.x * WAVES_PER_WG + wave;
    const int wstride  = gridDim.x * WAVES_PER_WG;

    for (int blk = wave_gid; blk < nblk; blk += wstride) {
        // ---- load 16x64 x-block directly in A-fragment order ----
        // lane holds row (nsel), K pattern {g*16+kbA .. +7} for g=0..3
        const float* xr = x + (size_t)blk * 1024 + (size_t)nsel * 64 + kbA;
        if (blk + wstride < nblk)
            __builtin_prefetch(x + (size_t)(blk + wstride) * 1024 + (size_t)nsel * 64 + kbA, 0, 0);
        v4f f[8];
        #pragma unroll
        for (int g = 0; g < 4; ++g) {
            f[2 * g]     = *(const v4f*)(xr + g * 16);
            f[2 * g + 1] = *(const v4f*)(xr + g * 16 + 4);
        }

        // ---- row norm: 32 local squares + partner-lane (xor 16) reduce ----
        float ss = 0.0f;
        #pragma unroll
        for (int i = 0; i < 8; ++i)
            #pragma unroll
            for (int j = 0; j < 4; ++j) ss += f[i][j] * f[i][j];
        ss += __shfl_xor(ss, 16, 32);
        float scale = fmaxf(sqrtf(ss), 1e-8f);
        float rinv  = 1.0f / scale;
        if (hi == 0) sc[wave][nsel] = scale;

        // ---- A fragments: normalized x, f16 ----
        v16h A0, A1;
        #pragma unroll
        for (int e = 0; e < 16; ++e) {
            A0[e] = (_Float16)(f[e >> 2][e & 3] * rinv);
            A1[e] = (_Float16)(f[(e + 16) >> 2][e & 3] * rinv);
        }

        // ---- matmul1 (z = x_hat @ R^T) + Lloyd-Max quantize per N-tile ----
        #pragma unroll
        for (int nt = 0; nt < 4; ++nt) {
            v8f acc = {};
            acc = wmma_f16(A0, B1[nt][0], acc);
            acc = wmma_f16(A1, B1[nt][1], acc);
            const int col = nt * 16 + nsel;
            #pragma unroll
            for (int v = 0; v < 8; ++v) {
                float z = acc[v];
                float q = cb[0];
                #pragma unroll
                for (int l = 1; l < 16; ++l)
                    q = (z > mid[l - 1]) ? cb[l] : q;   // nearest of sorted codebook
                const int m = v + hi * 8;               // D row (C/D VGPR layout)
                zt[wave][m * 64 + col] = (_Float16)q;
            }
        }

        // DS ops are in-order per wave; wait + compiler barrier before re-reading
        asm volatile("s_wait_dscnt 0" ::: "memory");

        // ---- reload z_hat in A-fragment layout (LDS round-trip = transpose) ----
        const _Float16* zr = &zt[wave][nsel * 64 + kbA];
        v8h r0 = *(const v8h*)(zr);
        v8h r1 = *(const v8h*)(zr + 16);
        v8h r2 = *(const v8h*)(zr + 32);
        v8h r3 = *(const v8h*)(zr + 48);
        v16h A2_0, A2_1;
        #pragma unroll
        for (int j = 0; j < 8; ++j) {
            A2_0[j] = r0[j]; A2_0[8 + j] = r1[j];
            A2_1[j] = r2[j]; A2_1[8 + j] = r3[j];
        }

        // per-row scales for this lane's 8 output rows
        v4f s0 = *(const v4f*)(&sc[wave][hi * 8]);
        v4f s1 = *(const v4f*)(&sc[wave][hi * 8 + 4]);

        // ---- matmul2 (x_rec = z_hat @ R), rescale, store ----
        #pragma unroll
        for (int nt = 0; nt < 4; ++nt) {
            v8f acc = {};
            acc = wmma_f16(A2_0, B2[nt][0], acc);
            acc = wmma_f16(A2_1, B2[nt][1], acc);
            const int col = nt * 16 + nsel;
            float* orow = out + (size_t)blk * 1024 + col;
            #pragma unroll
            for (int v = 0; v < 8; ++v) {
                const int m = v + hi * 8;
                const float s = (v < 4) ? s0[v] : s1[v - 4];
                orow[m * 64] = acc[v] * s;
            }
        }
    }
}

extern "C" void kernel_launch(void* const* d_in, const int* in_sizes, int n_in,
                              void* d_out, int out_size, void* d_ws, size_t ws_size,
                              hipStream_t stream) {
    const float* x  = (const float*)d_in[0];   // [B,H,S,64] f32
    const float* cb = (const float*)d_in[1];   // [16] f32
    const float* R  = (const float*)d_in[2];   // [64,64] f32
    float* out = (float*)d_out;

    const int rows = in_sizes[0] / 64;
    const int nblk = rows / 16;
    if (nblk <= 0) return;

    int wgs = (nblk + WAVES_PER_WG - 1) / WAVES_PER_WG;
    if (wgs > 1024) wgs = 1024;

    turboquant_kernel<<<dim3(wgs), dim3(256), 0, stream>>>(x, cb, R, out, nblk);
}
